// shallow_32796370272787
// MI455X (gfx1250) — matrix operand order repaired and checked by reading, not compiled, not run on hardware
//
#include <hip/hip_runtime.h>
#include <hip/hip_bf16.h>

// MI455X (gfx1250, wave32) implementation.
// conv2(1x1) o conv1(3x3) collapsed into one 64->8 3x3 conv via weight fusion,
// executed as implicit GEMM with v_wmma_f32_16x16x32_bf16.
// Conv kernel processes 16-row strips with a rolling 4-slab LDS ring so each
// input row is read from HBM ~once (1.125x amplification), keeping traffic
// near the 134 MB / 23.3 TB/s ~ 6 us roofline floor. Staging is strength-
// reduced: interior columns need no div/mod/bounds math (~5 instr/element).

typedef __attribute__((ext_vector_type(16))) __bf16 v16bf;
typedef __attribute__((ext_vector_type(8)))  __bf16 v8bf;
typedef __attribute__((ext_vector_type(8)))  float  v8f;

#define BATCH 8
#define CIN   64
#define CF    8
#define NBOX  4096
#define HH    256
#define WW    256
#define KSTEPS 18          // K = 9 positions * 64 ch = 576 -> 18 steps of 32
#define TILE_W 128
#define C_TILE 130         // 128 + 2 halo
#define CI_PAD 72          // padded channel dim in LDS (36 dwords/col stride)
#define STRIP  16          // output rows per block
#define SLAB   (C_TILE * CI_PAD)
#define OUT_TOT (BATCH*CF*NBOX*9)   // 2359296 per output tensor

// ws layout (bytes)
#define AFRAG_OFF 0                      // 18*32*16 bf16 = 18432 B
#define BEFF_OFF  18432                  // 8 floats
#define F2_OFF    20480                  // 8*8*256*256 f32 = 16 MiB

// A-fragment element->K mapping for 16-bit 16x32 A (ISA 7.12.2):
// lane<16 : elem i in [0,8) -> K = i      ; i in [8,16) -> K = 16 + (i-8)
// lane>=16: elem i in [0,8) -> K = 8 + i  ; i in [8,16) -> K = 24 + (i-8)
__device__ __forceinline__ int a_frag_k(int lane, int i) {
    return ((lane >= 16) ? 8 : 0) + ((i >= 8) ? 16 : 0) + (i & 7);
}

// ---------------------------------------------------------------------------
// Kernel 1: fuse w2 (1x1) into w1 (3x3): w_eff[cf,ci,kh,kw] = sum_c w2[cf,c]*w1[c,ci,kh,kw]
// Emit w_eff already in WMMA A-fragment order: Afrag[ks][lane][i], k = ks*32 + a_frag_k
// with global K ordering k = (kh*3+kw)*64 + ci. Rows M=8..15 zero-padded.
// Also b_eff[cf] = sum_c w2[cf,c]*b1[c].
// ---------------------------------------------------------------------------
__global__ void fuse_weights(const float* __restrict__ w1,
                             const float* __restrict__ b1,
                             const float* __restrict__ w2,
                             __bf16* __restrict__ Afrag,
                             float*  __restrict__ beff) {
    int t = blockIdx.x * blockDim.x + threadIdx.x;
    const int total = KSTEPS * 32 * 16;     // 9216
    for (int e = t; e < total; e += gridDim.x * blockDim.x) {
        int ks   = e >> 9;          // /512
        int lane = (e >> 4) & 31;
        int i    = e & 15;
        int m    = lane & 15;       // output channel row
        int k    = ks * 32 + a_frag_k(lane, i);
        int ci   = k & 63;
        int kpos = k >> 6;          // kh*3+kw
        float v = 0.f;
        if (m < CF) {
            #pragma unroll 8
            for (int c = 0; c < CIN; ++c)
                v += w2[m * CIN + c] * w1[(c * CIN + ci) * 9 + kpos];
        }
        Afrag[e] = (__bf16)v;
    }
    if (t < CF) {
        float v = 0.f;
        for (int c = 0; c < CIN; ++c) v += w2[t * CIN + c] * b1[c];
        beff[t] = v;
    }
}

// ---------------------------------------------------------------------------
// Kernel 2: fused 3x3 conv 64->8 as implicit GEMM with bf16 WMMA.
// Block = (batch b, half-row x0, 16-row strip); 8 waves, each 16cf x 16px.
// Rolling ring of 4 row-slabs in LDS: per output row, stage one new input
// row (issued before compute so the loads overlap the WMMA work).
// ---------------------------------------------------------------------------
__global__ __launch_bounds__(256) void conv_wmma(const float* __restrict__ feat,
                                                 const __bf16* __restrict__ Afrag,
                                                 const float* __restrict__ beff,
                                                 float* __restrict__ f2) {
    __shared__ __bf16 lds[4 * SLAB];   // 4 * 18720 B = 74880 B

    const int blk   = blockIdx.x;
    const int x0    = (blk & 1) * TILE_W;
    const int strip = (blk >> 1) & 15;
    const int b     = blk >> 5;
    const int y0    = strip * STRIP;
    const int tid   = threadIdx.x;

    const float* fbase = feat + (size_t)b * CIN * HH * WW;

    // Stage one input row (f32 -> bf16) into ring slab (yy & 3), layout [col][ci].
    // Interior 128 cols: no bounds checks, no div/mod; ci advances by 2 per step.
    auto stage_row = [&](int yy) {
        __bf16* slab = lds + (size_t)(yy & 3) * SLAB;
        if (yy < 0 || yy >= HH) {              // uniform branch: zero-fill row
            for (int j = tid; j < SLAB; j += 256) slab[j] = (__bf16)0.f;
        } else {
            const int c  = tid & 127;          // interior col 0..127 (cc = c+1)
            int ci       = tid >> 7;           // 0 or 1, then += 2
            const float* gp = fbase + ((size_t)ci * HH + yy) * WW + (x0 + c);
            int lofs = (c + 1) * CI_PAD + ci;
            #pragma unroll 8
            for (int it = 0; it < 32; ++it) {  // covers ci = 0..63
                float v = *gp;
                slab[lofs] = (__bf16)v;
                gp   += 2 * HH * WW;
                lofs += 2;
            }
            // halo cols cc=0 (xx=x0-1) and cc=129 (xx=x0+128): waves 0..3
            if (tid < 128) {
                int hci  = tid & 63;
                int side = tid >> 6;
                int cc   = side ? 129 : 0;
                int xx   = x0 - 1 + cc;
                float v = 0.f;
                if (xx >= 0 && xx < WW)
                    v = fbase[((size_t)hci * HH + yy) * WW + xx];
                slab[cc * CI_PAD + hci] = (__bf16)v;
            }
        }
    };

    stage_row(y0 - 1);
    stage_row(y0);
    stage_row(y0 + 1);
    __syncthreads();

    const int lane   = tid & 31;
    const int wave   = tid >> 5;
    const int n      = lane & 15;      // pixel within wave tile / output col
    const int laneHi = lane >> 4;      // which K/ci half of the fragment
    const int x      = x0 + wave * 16 + n;

    float bias[CF];
    #pragma unroll
    for (int r = 0; r < CF; ++r) bias[r] = beff[r];

    for (int y = y0; y < y0 + STRIP; ++y) {
        // Software pipeline: issue next row's global loads before compute.
        if (y + 2 <= y0 + STRIP) stage_row(y + 2);

        v8f acc = {};
        #pragma unroll
        for (int ks = 0; ks < KSTEPS; ++ks) {
            // A fragment: contiguous 32 B per lane, pre-layouted in scratch.
            v16bf a = *(const v16bf*)(Afrag + (size_t)ks * 512 + lane * 16);

            // B fragment: two runs of 8 channels (ci0, ci0+16) at (row y-1+kh, col) in LDS.
            const int kpos = ks >> 1;              // kh*3+kw
            const int chh  = ks & 1;               // channel half
            const int kh   = kpos / 3;
            const int kw   = kpos - kh * 3;
            const __bf16* slab = lds + (size_t)((y - 1 + kh) & 3) * SLAB;
            const int cc   = wave * 16 + n + kw;   // col in [0,130)
            const int ci0  = chh * 32 + laneHi * 8;
            const __bf16* p = slab + cc * CI_PAD + ci0;
            v8bf lo = *(const v8bf*)(p);
            v8bf hi = *(const v8bf*)(p + 16);
            v16bf bb;
            #pragma unroll
            for (int i = 0; i < 8; ++i) { bb[i] = lo[i]; bb[i + 8] = hi[i]; }

            acc = __builtin_amdgcn_wmma_f32_16x16x32_bf16(
                      false, a, false, bb, (short)0, acc, false, false);
        }

        // Lanes 0..15 hold output rows M=0..7 (all CF channels) in acc[0..7].
        if (lane < 16) {
            #pragma unroll
            for (int r = 0; r < CF; ++r)
                f2[(((size_t)b * CF + r) * HH + y) * WW + x] = acc[r] + bias[r];
        }
        __syncthreads();   // slab (y+2) visible; slab (y-1) free for reuse
    }
}

// ---------------------------------------------------------------------------
// Kernel 3: ROI align (OUT=3, S=2 samples) on L2-resident f2, fused with the
// attention multiply. One thread per output element.
// ---------------------------------------------------------------------------
__global__ __launch_bounds__(256) void roi_attn(const float* __restrict__ f2,
                                                const float* __restrict__ attn,
                                                const float* __restrict__ locat,
                                                float* __restrict__ out) {
    int idx = blockIdx.x * 256 + threadIdx.x;
    if (idx >= OUT_TOT) return;

    const int pix = idx % 9;
    const int box = idx / 9;
    const int ox  = pix % 3;
    const int oy  = pix / 3;
    const int bc  = box / NBOX;                 // plane index in [0, 64)

    const float* L = locat + (size_t)box * 4;
    const float x1 = L[0], y1 = L[1], x2 = L[2], y2 = L[3];
    const float sx = x1 - 0.5f, sy = y1 - 0.5f;
    const float bw = (x2 - x1) * (1.f / 3.f);
    const float bh = (y2 - y1) * (1.f / 3.f);

    const float* plane = f2 + (size_t)bc * HH * WW;

    float acc = 0.f;
    #pragma unroll
    for (int syi = 0; syi < 2; ++syi) {
        #pragma unroll
        for (int sxi = 0; sxi < 2; ++sxi) {
            float yf = sy + ((float)oy + ((float)syi + 0.5f) * 0.5f) * bh;
            float xf = sx + ((float)ox + ((float)sxi + 0.5f) * 0.5f) * bw;
            bool valid = (yf >= -1.f) && (yf <= (float)HH) &&
                         (xf >= -1.f) && (xf <= (float)WW);
            float yc = fminf(fmaxf(yf, 0.f), (float)(HH - 1));
            float xc = fminf(fmaxf(xf, 0.f), (float)(WW - 1));
            int iy0 = (int)floorf(yc);
            int ix0 = (int)floorf(xc);
            int iy1 = min(iy0 + 1, HH - 1);
            int ix1 = min(ix0 + 1, WW - 1);
            float ly = yc - (float)iy0, lx = xc - (float)ix0;
            float hy = 1.f - ly,        hx = 1.f - lx;
            float v = hy * hx * plane[iy0 * WW + ix0]
                    + hy * lx * plane[iy0 * WW + ix1]
                    + ly * hx * plane[iy1 * WW + ix0]
                    + ly * lx * plane[iy1 * WW + ix1];
            acc += valid ? v : 0.f;
        }
    }
    float patch = acc * 0.25f;
    out[idx]           = patch * attn[idx];   // output 0: patches * imp_attns
    out[OUT_TOT + idx] = patch;               // output 1: patches
}

// ---------------------------------------------------------------------------
extern "C" void kernel_launch(void* const* d_in, const int* in_sizes, int n_in,
                              void* d_out, int out_size, void* d_ws, size_t ws_size,
                              hipStream_t stream) {
    const float* features = (const float*)d_in[0];
    const float* imp_attn = (const float*)d_in[1];
    const float* imp_loc  = (const float*)d_in[2];
    const float* w1       = (const float*)d_in[3];
    const float* b1       = (const float*)d_in[4];
    const float* w2       = (const float*)d_in[5];

    char* ws = (char*)d_ws;
    __bf16* Afrag = (__bf16*)(ws + AFRAG_OFF);
    float*  beff  = (float*)(ws + BEFF_OFF);
    float*  f2    = (float*)(ws + F2_OFF);
    float*  out   = (float*)d_out;

    // 1) fuse conv2 into conv1, emit WMMA-layout A fragments
    fuse_weights<<<36, 256, 0, stream>>>(w1, b1, w2, Afrag, beff);

    // 2) fused 64->8 3x3 conv via bf16 WMMA: 8 b * 2 xhalf * 16 strips = 256 blocks
    conv_wmma<<<BATCH * 2 * (HH / STRIP), 256, 0, stream>>>(features, Afrag, beff, f2);

    // 3) ROI align + attention multiply
    roi_attn<<<(OUT_TOT + 255) / 256, 256, 0, stream>>>(f2, imp_attn, imp_loc, out);
}
